// ProtoMix_36550171689554
// MI455X (gfx1250) — matrix operand order repaired and checked by reading, not compiled, not run on hardware
//
#include <hip/hip_runtime.h>
#include <hip/hip_bf16.h>
#include <math.h>

typedef __bf16 bf16_t;
typedef __attribute__((ext_vector_type(16))) __bf16 v16bf;
typedef __attribute__((ext_vector_type(8)))  __bf16 v8bf;
typedef __attribute__((ext_vector_type(4)))  __bf16 v4bf;
typedef __attribute__((ext_vector_type(8)))  float  v8f;
typedef __attribute__((ext_vector_type(4)))  float  f32x4;

#define B_N   16384
#define D_N   1024
#define P_N   1024
#define OUT_N 25

static __device__ __forceinline__ v16bf cat8(v8bf lo, v8bf hi) {
  return __builtin_shufflevector(lo, hi, 0,1,2,3,4,5,6,7,8,9,10,11,12,13,14,15);
}

// ---------------- precision conversion ----------------
__global__ void cast_bf16_kernel(const float* __restrict__ src, bf16_t* __restrict__ dst, int n) {
  int i = blockIdx.x * blockDim.x + threadIdx.x;
  if (i < n) dst[i] = (bf16_t)src[i];
}

// dst[c*rows + r] = src[r*cols + c]  (store transpose, row-major [cols][rows])
__global__ void castT_bf16_kernel(const float* __restrict__ src, bf16_t* __restrict__ dst,
                                  int rows, int cols) {
  int i = blockIdx.x * blockDim.x + threadIdx.x;
  if (i < rows * cols) {
    int r = i / cols, c = i - r * cols;
    dst[(size_t)c * rows + r] = (bf16_t)src[i];
  }
}

// class_fc_w [25][1024] fp32 -> zero-padded transposed bf16 B-panel [1024][32]
__global__ void cast_clsT_kernel(const float* __restrict__ cls_w, bf16_t* __restrict__ dst) {
  int i = blockIdx.x * blockDim.x + threadIdx.x;
  if (i < P_N * 32) {
    int k = i >> 5, o = i & 31;
    float v = (o < OUT_N) ? cls_w[(size_t)o * P_N + k] : 0.f;
    dst[i] = (bf16_t)v;
  }
}

// ---------------- per-prototype row stats: |p|^2, la, lb ----------------
__global__ void __launch_bounds__(128)
rowstats_kernel(const float* __restrict__ protos, const float* __restrict__ lin_w,
                float* __restrict__ p_sq, float* __restrict__ la, float* __restrict__ lb) {
  const int j = blockIdx.x;
  const int t = threadIdx.x;
  const float* row = protos + (size_t)j * D_N;
  float s = 0.f, a = 0.f, b = 0.f;
  for (int i = t; i < D_N; i += 128) {
    float x = row[i];
    s += x * x;
    a += x * lin_w[i];
    b += x * lin_w[D_N + i];
  }
  for (int m = 16; m >= 1; m >>= 1) {
    s += __shfl_xor(s, m, 32); a += __shfl_xor(a, m, 32); b += __shfl_xor(b, m, 32);
  }
  __shared__ float sh[3][4];
  const int wv = t >> 5, lane = t & 31;
  if (lane == 0) { sh[0][wv] = s; sh[1][wv] = a; sh[2][wv] = b; }
  __syncthreads();
  if (t == 0) {
    p_sq[j] = sh[0][0] + sh[0][1] + sh[0][2] + sh[0][3];
    la[j]   = sh[1][0] + sh[1][1] + sh[1][2] + sh[1][3];
    lb[j]   = sh[2][0] + sh[2][1] + sh[2][2] + sh[2][3];
  }
}

// ---------------- WMMA GEMM + fused argmin -> labels_a ----------------
// 16 patient rows per block; 4 waves sweep 64 N-tiles; argmin of (|p|^2 - 2 e.p)
__global__ void __launch_bounds__(128)
labels_kernel(const float* __restrict__ patient, const bf16_t* __restrict__ protosT,
              const float* __restrict__ p_sq, int* __restrict__ labels) {
  __shared__ __align__(16) bf16_t Ash[16 * D_N];   // 32 KB, bf16 A tile
  __shared__ float redV[4][16];
  __shared__ int   redI[4][16];
  const int blk  = blockIdx.x;
  const int tid  = threadIdx.x;
  const int lane = tid & 31, wv = tid >> 5;
  const int m = lane & 15, hi = lane >> 4;

  // stage A tile: fp32 -> bf16 into LDS (vectorized)
  const f32x4* Abase = (const f32x4*)(patient + (size_t)blk * 16 * D_N);
  for (int e4 = tid; e4 < 16 * D_N / 4; e4 += 128) {
    f32x4 f = Abase[e4];
    v4bf h;
#pragma unroll
    for (int j = 0; j < 4; ++j) h[j] = (bf16_t)f[j];
    ((v4bf*)Ash)[e4] = h;
  }
  __syncthreads();

  float minv[8]; int mini[8];
#pragma unroll
  for (int r = 0; r < 8; ++r) { minv[r] = 3.0e38f; mini[r] = 0; }

  for (int nt = wv; nt < P_N / 16; nt += 4) {
    v8f c = {};
    const bf16_t* Bbase = protosT + (size_t)nt * 16;
#pragma unroll 4
    for (int k0 = 0; k0 < D_N; k0 += 32) {
      v8bf alo = *(const v8bf*)&Ash[m * D_N + k0 + hi * 8];
      v8bf ahi = *(const v8bf*)&Ash[m * D_N + k0 + 16 + hi * 8];
      v16bf a  = cat8(alo, ahi);
      v16bf bm;
#pragma unroll
      for (int s = 0; s < 16; ++s) bm[s] = Bbase[(size_t)(k0 + (s & 8) * 2 + hi * 8 + (s & 7)) * P_N + m];
      c = __builtin_amdgcn_wmma_f32_16x16x32_bf16(false, a, false, bm, (short)0, c, false, false);
    }
    float psq = p_sq[nt * 16 + m];
#pragma unroll
    for (int r = 0; r < 8; ++r) {
      float s = psq - 2.0f * c[r];                 // argmin of d^2 (sqrt & |e|^2 dropped)
      if (s < minv[r]) { minv[r] = s; mini[r] = nt * 16 + m; }
    }
  }
  // butterfly min over the 16 lanes of each half (columns of the C tile)
#pragma unroll
  for (int msk = 1; msk <= 8; msk <<= 1) {
#pragma unroll
    for (int r = 0; r < 8; ++r) {
      float ov = __shfl_xor(minv[r], msk, 32);
      int   oi = __shfl_xor(mini[r], msk, 32);
      if (ov < minv[r]) { minv[r] = ov; mini[r] = oi; }
    }
  }
  if (m == 0) {
#pragma unroll
    for (int r = 0; r < 8; ++r) { redV[wv][r + hi * 8] = minv[r]; redI[wv][r + hi * 8] = mini[r]; }
  }
  __syncthreads();
  if (tid < 16) {
    float bv = redV[0][tid]; int bi = redI[0][tid];
    for (int w = 1; w < 4; ++w)
      if (redV[w][tid] < bv) { bv = redV[w][tid]; bi = redI[w][tid]; }
    labels[blk * 16 + tid] = bi;
  }
}

// ---------------- V = protos @ W  (bf16 WMMA, P x D, K = D) ----------------
__global__ void __launch_bounds__(128)
vmat_kernel(const bf16_t* __restrict__ protos_bf, const bf16_t* __restrict__ w_bf,
            bf16_t* __restrict__ v_bf) {
  const int mt   = blockIdx.x;
  const int tid  = threadIdx.x;
  const int lane = tid & 31, wv = tid >> 5;
  const int m = lane & 15, hi = lane >> 4;
  const bf16_t* Arow = protos_bf + (size_t)(mt * 16 + m) * D_N;
  for (int nt = wv; nt < D_N / 16; nt += 4) {
    v8f c = {};
#pragma unroll 4
    for (int k0 = 0; k0 < D_N; k0 += 32) {
      v8bf alo = *(const v8bf*)(Arow + k0 + hi * 8);
      v8bf ahi = *(const v8bf*)(Arow + k0 + 16 + hi * 8);
      v16bf a  = cat8(alo, ahi);
      v16bf bm;
#pragma unroll
      for (int s = 0; s < 16; ++s) bm[s] = w_bf[(size_t)(k0 + (s & 8) * 2 + hi * 8 + (s & 7)) * D_N + nt * 16 + m];
      c = __builtin_amdgcn_wmma_f32_16x16x32_bf16(false, a, false, bm, (short)0, c, false, false);
    }
#pragma unroll
    for (int r = 0; r < 8; ++r)
      v_bf[(size_t)(mt * 16 + r + hi * 8) * D_N + nt * 16 + m] = (bf16_t)c[r];
  }
}

// ---------------- lam[b] = sigmoid(V[ja].protos[jb] + la[ja] + lb[jb] + b0) ----------------
__global__ void __launch_bounds__(256)
lam_kernel(const int* __restrict__ labels, const int* __restrict__ index,
           const bf16_t* __restrict__ v_bf, const bf16_t* __restrict__ protos_bf,
           const float* __restrict__ la, const float* __restrict__ lb,
           const float* __restrict__ bil_b, float* __restrict__ lam) {
  const int tid  = threadIdx.x;
  const int lane = tid & 31, wv = tid >> 5;
  const int b  = blockIdx.x * 8 + wv;
  const int ja = labels[b];
  const int jb = labels[index[b]];             // labels_b = labels_a[index]
  const v8bf* vr = (const v8bf*)(v_bf      + (size_t)ja * D_N + lane * 32);
  const v8bf* pr = (const v8bf*)(protos_bf + (size_t)jb * D_N + lane * 32);
  float acc = 0.f;
#pragma unroll
  for (int c = 0; c < 4; ++c) {
    v8bf x = vr[c], y = pr[c];
#pragma unroll
    for (int j = 0; j < 8; ++j) acc += (float)x[j] * (float)y[j];
  }
  for (int msk = 16; msk >= 1; msk >>= 1) acc += __shfl_xor(acc, msk, 32);
  if (lane == 0) {
    float z = acc + la[ja] + lb[jb] + bil_b[0];
    lam[b] = 1.0f / (1.0f + __expf(-z));
  }
}

// ---------------- fused: mixup staging + proto_fc WMMA + leakyReLU + WMMA class head ----------------
__global__ void __launch_bounds__(128)
final_kernel(const float* __restrict__ patient, const int* __restrict__ index,
             const float* __restrict__ lam, const bf16_t* __restrict__ pfcT,
             const float* __restrict__ pfc_b, const bf16_t* __restrict__ clsT,
             const float* __restrict__ cls_b, float* __restrict__ out) {
  __shared__ __align__(16) bf16_t Ash[16 * D_N];   // 32 KB mixed-embed tile
  __shared__ __align__(16) bf16_t PLsh[16 * P_N];  // 32 KB leaky-ReLU'd proto logits
  const int blk  = blockIdx.x;
  const int tid  = threadIdx.x;
  const int lane = tid & 31, wv = tid >> 5;
  const int m = lane & 15, hi = lane >> 4;

  // stage mixed rows: lam*e + (1-lam)*e[index]
  for (int e4 = tid; e4 < 16 * D_N / 4; e4 += 128) {
    int row = e4 >> 8;                    // (e4*4) / 1024
    int bg  = blk * 16 + row;
    float l = lam[bg];
    int  pi = index[bg];
    f32x4 a  = ((const f32x4*)(patient + (size_t)bg * D_N))[e4 & 255];
    f32x4 bb = ((const f32x4*)(patient + (size_t)pi * D_N))[e4 & 255];
    v4bf h;
#pragma unroll
    for (int j = 0; j < 4; ++j) h[j] = (bf16_t)(l * a[j] + (1.0f - l) * bb[j]);
    ((v4bf*)Ash)[e4] = h;
  }
  __syncthreads();

  // proto_fc GEMM: PL = leaky_relu(mixed @ pfc^T + b)
  for (int nt = wv; nt < P_N / 16; nt += 4) {
    v8f c = {};
#pragma unroll 4
    for (int k0 = 0; k0 < D_N; k0 += 32) {
      v8bf alo = *(const v8bf*)&Ash[m * D_N + k0 + hi * 8];
      v8bf ahi = *(const v8bf*)&Ash[m * D_N + k0 + 16 + hi * 8];
      v16bf a  = cat8(alo, ahi);
      v16bf bm;
#pragma unroll
      for (int s = 0; s < 16; ++s) bm[s] = pfcT[(size_t)(k0 + (s & 8) * 2 + hi * 8 + (s & 7)) * P_N + nt * 16 + m];
      c = __builtin_amdgcn_wmma_f32_16x16x32_bf16(false, a, false, bm, (short)0, c, false, false);
    }
    float bias = pfc_b[nt * 16 + m];
#pragma unroll
    for (int r = 0; r < 8; ++r) {
      float x = c[r] + bias;
      x = (x > 0.f) ? x : 0.01f * x;               // leaky_relu
      PLsh[(r + hi * 8) * P_N + nt * 16 + m] = (bf16_t)x;
    }
  }
  __syncthreads();

  // class head as WMMA: out[16][25] = PL[16][1024] @ clsT (zero-padded to N=32)
  if (wv < 2) {
    const int nt = wv;                               // two 16-col tiles cover 32 >= 25
    v8f c = {};
#pragma unroll 4
    for (int k0 = 0; k0 < P_N; k0 += 32) {
      v8bf alo = *(const v8bf*)&PLsh[m * P_N + k0 + hi * 8];
      v8bf ahi = *(const v8bf*)&PLsh[m * P_N + k0 + 16 + hi * 8];
      v16bf a  = cat8(alo, ahi);
      v16bf bm;
#pragma unroll
      for (int s = 0; s < 16; ++s) bm[s] = clsT[(size_t)(k0 + (s & 8) * 2 + hi * 8 + (s & 7)) * 32 + nt * 16 + m];
      c = __builtin_amdgcn_wmma_f32_16x16x32_bf16(false, a, false, bm, (short)0, c, false, false);
    }
    const int n = nt * 16 + m;
    if (n < OUT_N) {
      float bias = cls_b[n];
#pragma unroll
      for (int r = 0; r < 8; ++r)
        out[(size_t)(blk * 16 + r + hi * 8) * OUT_N + n] = c[r] + bias;
    }
  }
}

// ---------------- host launch ----------------
extern "C" void kernel_launch(void* const* d_in, const int* in_sizes, int n_in,
                              void* d_out, int out_size, void* d_ws, size_t ws_size,
                              hipStream_t stream) {
  (void)in_sizes; (void)n_in; (void)out_size; (void)ws_size;
  const float* patient = (const float*)d_in[0];
  const float* protos  = (const float*)d_in[1];
  const int*   index   = (const int*)d_in[2];
  const float* bil_w   = (const float*)d_in[3];
  const float* bil_b   = (const float*)d_in[4];
  const float* lin_w   = (const float*)d_in[5];
  const float* pfc_w   = (const float*)d_in[6];
  const float* pfc_b   = (const float*)d_in[7];
  const float* cls_w   = (const float*)d_in[8];
  const float* cls_b   = (const float*)d_in[9];
  float* out = (float*)d_out;

  char* ws = (char*)d_ws;
  size_t off = 0;
  bf16_t* protos_bf = (bf16_t*)(ws + off); off += (size_t)P_N * D_N * 2;
  bf16_t* protosT   = (bf16_t*)(ws + off); off += (size_t)D_N * P_N * 2;
  bf16_t* w_bf      = (bf16_t*)(ws + off); off += (size_t)D_N * D_N * 2;
  bf16_t* pfcT      = (bf16_t*)(ws + off); off += (size_t)D_N * P_N * 2;
  bf16_t* v_bf      = (bf16_t*)(ws + off); off += (size_t)P_N * D_N * 2;
  bf16_t* clsT      = (bf16_t*)(ws + off); off += (size_t)P_N * 32 * 2;
  float*  p_sq      = (float*)(ws + off);  off += P_N * 4;
  float*  la        = (float*)(ws + off);  off += P_N * 4;
  float*  lb        = (float*)(ws + off);  off += P_N * 4;
  int*    labels    = (int*)(ws + off);    off += B_N * 4;
  float*  lam       = (float*)(ws + off);  off += B_N * 4;

  const int n1 = P_N * D_N;
  cast_bf16_kernel <<<(n1 + 255) / 256, 256, 0, stream>>>(protos, protos_bf, n1);
  castT_bf16_kernel<<<(n1 + 255) / 256, 256, 0, stream>>>(protos, protosT, P_N, D_N);
  cast_bf16_kernel <<<(n1 + 255) / 256, 256, 0, stream>>>(bil_w, w_bf, D_N * D_N);
  castT_bf16_kernel<<<(n1 + 255) / 256, 256, 0, stream>>>(pfc_w, pfcT, P_N, D_N);
  cast_clsT_kernel <<<(P_N * 32 + 255) / 256, 256, 0, stream>>>(cls_w, clsT);
  rowstats_kernel  <<<P_N, 128, 0, stream>>>(protos, lin_w, p_sq, la, lb);
  labels_kernel    <<<B_N / 16, 128, 0, stream>>>(patient, protosT, p_sq, labels);
  vmat_kernel      <<<P_N / 16, 128, 0, stream>>>(protos_bf, w_bf, v_bf);
  lam_kernel       <<<B_N / 8, 256, 0, stream>>>(labels, index, v_bf, protos_bf, la, lb, bil_b, lam);
  final_kernel     <<<B_N / 16, 128, 0, stream>>>(patient, index, lam, pfcT, pfc_b, clsT, cls_b, out);
}
